// GatedDeltaNet_695784702658
// MI455X (gfx1250) — compile-verified, matrix operand order checked by reading, not run on hardware
//
#include <hip/hip_runtime.h>
#include <hip/hip_bf16.h>
#include <math.h>

typedef __attribute__((ext_vector_type(16))) __bf16 v16bf;
typedef __attribute__((ext_vector_type(8)))  float  v8f;

namespace gdn {
constexpr int Bn   = 2;
constexpr int Tn   = 2048;
constexpr int Dn   = 2048;
constexpr int Hn   = 16;
constexpr int DKn  = 96;
constexpr int DVn  = 192;
constexpr int KEY  = Hn * DKn;                        // 1536
constexpr int VAL  = Hn * DVn;                        // 3072
constexpr int Mn   = Bn * Tn;                         // 4096 rows
constexpr int NC   = KEY + KEY + VAL + VAL + Hn + Hn; // 9248 concat cols
constexpr int NCP  = 9472;                            // padded to multiple of 256
constexpr int KSn  = 4;
constexpr int COL_Q = 0, COL_K = 1536, COL_V = 3072, COL_G = 6144;
constexpr int COL_B = 9216, COL_A = 9232;

// Fragment-packed element index (bf16 WMMA 16x16x32 operand layouts).
// Block (tile16, kblock32) -> 512 contiguous bf16: lane*16 + e.
__host__ __device__ inline size_t pk_index_b(int kk, int n, int NT) {
  int nt = n >> 4, l15 = n & 15;
  int kb = kk >> 5, kloc = kk & 31;
  int hi = (kloc >> 3) & 1;
  int e  = (kloc < 16) ? (kloc - hi * 8) : (kloc - 8 - hi * 8);
  int lane = l15 + (hi << 4);
  return ((size_t)kb * NT + nt) * 512 + (size_t)lane * 16 + e;
}
__host__ __device__ inline size_t pk_index_a(int m, int kk, int KB) {
  int mb = m >> 4, l15 = m & 15;
  int kb = kk >> 5, kloc = kk & 31;
  int hi = (kloc >> 3) & 1;
  int e  = (kloc < 16) ? (kloc - hi * 8) : (kloc - 8 - hi * 8);
  int lane = l15 + (hi << 4);
  return ((size_t)mb * KB + kb) * 512 + (size_t)lane * 16 + e;
}
}

// ---------------------------------------------------------------------------
// Pack fp32 row-major A [M][K] -> bf16 WMMA-fragment layout
// ---------------------------------------------------------------------------
__global__ void k_pack_a(const float* __restrict__ src, __bf16* __restrict__ dst,
                         int Mr, int K) {
  using namespace gdn;
  size_t i = (size_t)blockIdx.x * blockDim.x + threadIdx.x;
  if (i >= (size_t)Mr * K) return;
  int m  = (int)(i / K);
  int kk = (int)(i % K);
  dst[pk_index_a(m, kk, K >> 5)] = (__bf16)src[i];
}

// ---------------------------------------------------------------------------
// Build packed concatenated weights [D][NCP]: [Wq|Wk|Wv|Wg|Wb|Wa|pad0]
// ---------------------------------------------------------------------------
__global__ void k_pack_wcat(const float* __restrict__ Wq, const float* __restrict__ Wk,
                            const float* __restrict__ Wv, const float* __restrict__ Wg,
                            const float* __restrict__ Wb, const float* __restrict__ Wa,
                            __bf16* __restrict__ out) {
  using namespace gdn;
  size_t i = (size_t)blockIdx.x * blockDim.x + threadIdx.x;
  if (i >= (size_t)Dn * NCP) return;
  int kk = (int)(i / NCP);
  int n  = (int)(i % NCP);
  float v = 0.f;
  if      (n < COL_K) v = Wq[(size_t)kk * KEY + n];
  else if (n < COL_V) v = Wk[(size_t)kk * KEY + (n - COL_K)];
  else if (n < COL_G) v = Wv[(size_t)kk * VAL + (n - COL_V)];
  else if (n < COL_B) v = Wg[(size_t)kk * VAL + (n - COL_G)];
  else if (n < COL_A) v = Wb[(size_t)kk * Hn + (n - COL_B)];
  else if (n < NC)    v = Wa[(size_t)kk * Hn + (n - COL_A)];
  out[pk_index_b(kk, n, NCP >> 4)] = (__bf16)v;
}

// ---------------------------------------------------------------------------
// Pack Wo [VAL][Dn] -> bf16 fragment layout
// ---------------------------------------------------------------------------
__global__ void k_pack_wo(const float* __restrict__ Wo, __bf16* __restrict__ out) {
  using namespace gdn;
  size_t i = (size_t)blockIdx.x * blockDim.x + threadIdx.x;
  if (i >= (size_t)VAL * Dn) return;
  int kk = (int)(i / Dn);
  int n  = (int)(i % Dn);
  out[pk_index_b(kk, n, Dn >> 4)] = (__bf16)Wo[i];
}

// ---------------------------------------------------------------------------
// WMMA bf16 GEMM on fragment-packed operands, software-pipelined.
// Wave tile: 32M x 64N (2 A-frags x 4 B-frags, 8 WMMA per K-step).
// Block: 256 threads = 8 waves arranged 2(M) x 4(N) -> 64M x 256N.
// Fragment loads: one contiguous 32B read per lane (2x b128).
// Next K-step's fragments are issued before the current step's WMMAs so the
// load wait is covered by 8 matrix ops instead of stalling.
// ---------------------------------------------------------------------------
__device__ __forceinline__ v8f wmma_bf16(v16bf a, v16bf b, v8f c) {
  return __builtin_amdgcn_wmma_f32_16x16x32_bf16(false, a, false, b, (short)0, c,
                                                 false, false);
}

__global__ void __launch_bounds__(256)
k_wmma_gemm_pk(const __bf16* __restrict__ Apk, const __bf16* __restrict__ Bpk,
               float* __restrict__ C, int ldc, int Nreal, int K, int NT) {
  int wave = threadIdx.x >> 5;
  int lane = threadIdx.x & 31;
  int wm = wave >> 2, wn = wave & 3;
  int mb0 = blockIdx.x * 4 + wm * 2;     // first 16-row tile (2 tiles per wave)
  int nt0 = blockIdx.y * 16 + wn * 4;    // first 16-col tile (4 tiles per wave)
  int KB  = K >> 5;

  const __bf16* a0p = Apk + ((size_t)(mb0 + 0) * KB) * 512 + (size_t)lane * 16;
  const __bf16* a1p = Apk + ((size_t)(mb0 + 1) * KB) * 512 + (size_t)lane * 16;
  const __bf16* bp  = Bpk + ((size_t)nt0 * 512) + (size_t)lane * 16;
  const size_t bstride = (size_t)NT * 512;

  v8f acc[2][4] = {};

  // prologue: load kb = 0
  v16bf a0c = *(const v16bf*)a0p;
  v16bf a1c = *(const v16bf*)a1p;
  v16bf bc[4];
#pragma unroll
  for (int j = 0; j < 4; ++j) bc[j] = *(const v16bf*)(bp + (size_t)j * 512);

#pragma unroll 2
  for (int kb = 0; kb + 1 < KB; ++kb) {
    // issue next-step loads first
    v16bf a0n = *(const v16bf*)(a0p + (size_t)(kb + 1) * 512);
    v16bf a1n = *(const v16bf*)(a1p + (size_t)(kb + 1) * 512);
    const __bf16* brow = bp + (size_t)(kb + 1) * bstride;
    v16bf bn[4];
#pragma unroll
    for (int j = 0; j < 4; ++j) bn[j] = *(const v16bf*)(brow + (size_t)j * 512);

    if (kb + 8 < KB) {
      __builtin_prefetch(a0p + (size_t)(kb + 8) * 512, 0, 3);
      __builtin_prefetch(a1p + (size_t)(kb + 8) * 512, 0, 3);
      __builtin_prefetch(bp + (size_t)(kb + 8) * bstride, 0, 3);
      __builtin_prefetch(bp + (size_t)(kb + 8) * bstride + 1024, 0, 3);
    }

    // compute current step
#pragma unroll
    for (int j = 0; j < 4; ++j) {
      acc[0][j] = wmma_bf16(a0c, bc[j], acc[0][j]);
      acc[1][j] = wmma_bf16(a1c, bc[j], acc[1][j]);
    }
    a0c = a0n; a1c = a1n;
#pragma unroll
    for (int j = 0; j < 4; ++j) bc[j] = bn[j];
  }

  // epilogue: last step
#pragma unroll
  for (int j = 0; j < 4; ++j) {
    acc[0][j] = wmma_bf16(a0c, bc[j], acc[0][j]);
    acc[1][j] = wmma_bf16(a1c, bc[j], acc[1][j]);
  }

  int l15 = lane & 15, hi = lane >> 4;
#pragma unroll
  for (int i = 0; i < 2; ++i) {
#pragma unroll
    for (int j = 0; j < 4; ++j) {
      int nn = (nt0 + j) * 16 + l15;
      if (nn < Nreal) {
#pragma unroll
        for (int r = 0; r < 8; ++r) {
          int mm = (mb0 + i) * 16 + r + hi * 8;
          C[(size_t)mm * ldc + nn] = acc[i][j][r];
        }
      }
    }
  }
}

// ---------------------------------------------------------------------------
// Depthwise causal conv (KS=4) + SiLU over q|k|v channels of proj.
// ---------------------------------------------------------------------------
__global__ void k_conv_silu(const float* __restrict__ proj,
                            const float* __restrict__ cwq,
                            const float* __restrict__ cwk,
                            const float* __restrict__ cwv,
                            float* __restrict__ qc,
                            float* __restrict__ kc,
                            float* __restrict__ vc) {
  using namespace gdn;
  size_t i = (size_t)blockIdx.x * blockDim.x + threadIdx.x;
  constexpr int CC = KEY + KEY + VAL;  // 6144
  if (i >= (size_t)Mn * CC) return;
  int mrow = (int)(i / CC);
  int cc   = (int)(i % CC);
  int t = mrow % Tn;

  int col; const float* w; float* dst; int dld; int dcol;
  if (cc < KEY)          { col = COL_Q + cc;     w = cwq + (size_t)cc * KSn;       dst = qc; dld = KEY; dcol = cc; }
  else if (cc < 2 * KEY) { int c = cc - KEY;     w = cwk + (size_t)c * KSn; col = COL_K + c; dst = kc; dld = KEY; dcol = c; }
  else                   { int c = cc - 2 * KEY; w = cwv + (size_t)c * KSn; col = COL_V + c; dst = vc; dld = VAL; dcol = c; }

  float y = 0.f;
#pragma unroll
  for (int j = 0; j < KSn; ++j) {
    int tt = t - (KSn - 1) + j;
    if (tt >= 0)
      y += proj[(size_t)(mrow - (KSn - 1) + j) * NC + col] * w[j];
  }
  y = y / (1.f + expf(-y));  // SiLU
  dst[(size_t)mrow * dld + dcol] = y;
}

// ---------------------------------------------------------------------------
// beta = sigmoid(x@Wb);  g = -exp(A_log) * softplus(x@Wa + dt_bias)
// ---------------------------------------------------------------------------
__global__ void k_beta_gate(const float* __restrict__ proj,
                            const float* __restrict__ A_log,
                            const float* __restrict__ dt_bias,
                            float* __restrict__ beta,
                            float* __restrict__ gdec) {
  using namespace gdn;
  size_t i = (size_t)blockIdx.x * blockDim.x + threadIdx.x;
  if (i >= (size_t)Mn * Hn) return;
  int mrow = (int)(i / Hn);
  int h    = (int)(i % Hn);
  float xb = proj[(size_t)mrow * NC + COL_B + h];
  beta[i] = 1.f / (1.f + expf(-xb));
  float xa = proj[(size_t)mrow * NC + COL_A + h] + dt_bias[h];
  float sp = (xa > 20.f) ? xa : log1pf(expf(xa));
  gdec[i] = -expf(A_log[h]) * sp;
}

// ---------------------------------------------------------------------------
// Per-(row, head) L2 normalization of q and k (in place).
// ---------------------------------------------------------------------------
__global__ void k_l2norm_qk(float* __restrict__ qc, float* __restrict__ kc) {
  using namespace gdn;
  size_t i = (size_t)blockIdx.x * blockDim.x + threadIdx.x;
  if (i >= (size_t)Mn * Hn * 2) return;
  int which = (int)(i & 1);
  size_t mh = i >> 1;
  int mrow = (int)(mh / Hn);
  int h    = (int)(mh % Hn);
  float* p = (which ? kc : qc) + (size_t)mrow * KEY + (size_t)h * DKn;
  float s = 0.f;
#pragma unroll 4
  for (int d = 0; d < DKn; ++d) s += p[d] * p[d];
  float r = rsqrtf(s + 1e-12f);
#pragma unroll 4
  for (int d = 0; d < DKn; ++d) p[d] *= r;
}

// ---------------------------------------------------------------------------
// Gated delta-rule scan. One block per (b,h); 192 lanes each own one DV
// column of the 96x192 state, held entirely in VGPRs. k/q broadcast via LDS.
// ---------------------------------------------------------------------------
__global__ void __launch_bounds__(192)
k_delta_scan(const float* __restrict__ qc, const float* __restrict__ kc,
             const float* __restrict__ vc, const float* __restrict__ beta,
             const float* __restrict__ gdec, float* __restrict__ out) {
  using namespace gdn;
  int bh = blockIdx.x;
  int b = bh / Hn, h = bh % Hn;
  int tid = threadIdx.x;            // 0..191 = DV column

  __shared__ float kq[2 * DKn];     // k in [0,96), q in [96,192)

  float S[DKn];
#pragma unroll
  for (int d = 0; d < DKn; ++d) S[d] = 0.f;

  for (int t = 0; t < Tn; ++t) {
    int mrow = b * Tn + t;
    __syncthreads();
    if (tid < DKn)
      kq[tid] = kc[(size_t)mrow * KEY + (size_t)h * DKn + tid];
    else
      kq[tid] = qc[(size_t)mrow * KEY + (size_t)h * DKn + (tid - DKn)];
    __syncthreads();

    float alpha = expf(gdec[(size_t)mrow * Hn + h]);
    float bta   = beta[(size_t)mrow * Hn + h];
    float vv    = vc[(size_t)mrow * VAL + (size_t)h * DVn + tid];

    float r = 0.f, p = 0.f, qk = 0.f;
#pragma unroll
    for (int d = 0; d < DKn; ++d) {
      float kkv = kq[d], qqv = kq[DKn + d], s = S[d];
      r  += kkv * s;
      p  += qqv * s;
      qk += kkv * qqv;
    }
    float err = vv - r;
    // o_t = q . S_new = alpha*(q.S_old) + beta*(q.k)*err
    float o = alpha * p + bta * qk * err;
    float be = bta * err;
#pragma unroll
    for (int d = 0; d < DKn; ++d) S[d] = alpha * S[d] + kq[d] * be;

    out[(size_t)mrow * VAL + (size_t)h * DVn + tid] = o;
  }
}

// ---------------------------------------------------------------------------
// Gated RMSNorm over DV + sigmoid gate; emit bf16 directly in A-fragment
// packed layout for the final GEMM (K dimension = VAL channel).
// ---------------------------------------------------------------------------
__global__ void k_rms_gate(const float* __restrict__ o,
                           const float* __restrict__ proj,
                           const float* __restrict__ wnorm,
                           __bf16* __restrict__ obpk) {
  using namespace gdn;
  size_t i = (size_t)blockIdx.x * blockDim.x + threadIdx.x;
  if (i >= (size_t)Mn * Hn) return;
  int mrow = (int)(i / Hn);
  int h    = (int)(i % Hn);
  const float* op = o + (size_t)mrow * VAL + (size_t)h * DVn;
  float s = 0.f;
#pragma unroll 4
  for (int v = 0; v < DVn; ++v) s += op[v] * op[v];
  float rms = rsqrtf(s * (1.f / DVn) + 1e-6f);
#pragma unroll 4
  for (int v = 0; v < DVn; ++v) {
    float gp = proj[(size_t)mrow * NC + COL_G + (size_t)h * DVn + v];
    float val = op[v] * rms * wnorm[v] * (1.f / (1.f + expf(-gp)));
    int kk = h * DVn + v;
    obpk[pk_index_a(mrow, kk, VAL >> 5)] = (__bf16)val;
  }
}

// ---------------------------------------------------------------------------
// launch
// ---------------------------------------------------------------------------
extern "C" void kernel_launch(void* const* d_in, const int* in_sizes, int n_in,
                              void* d_out, int out_size, void* d_ws, size_t ws_size,
                              hipStream_t stream) {
  using namespace gdn;
  const float* x    = (const float*)d_in[0];
  const float* Wq   = (const float*)d_in[1];
  const float* Wk   = (const float*)d_in[2];
  const float* Wv   = (const float*)d_in[3];
  const float* Wb   = (const float*)d_in[4];
  const float* Wa   = (const float*)d_in[5];
  const float* Wg   = (const float*)d_in[6];
  const float* Wo   = (const float*)d_in[7];
  const float* cwq  = (const float*)d_in[8];
  const float* cwk  = (const float*)d_in[9];
  const float* cwv  = (const float*)d_in[10];
  const float* Alog = (const float*)d_in[11];
  const float* dtb  = (const float*)d_in[12];
  const float* wnrm = (const float*)d_in[13];
  float* out = (float*)d_out;

  char* ws = (char*)d_ws;
  size_t off = 0;
  auto alloc = [&](size_t bytes) {
    void* p = ws + off;
    off += (bytes + 255) & ~(size_t)255;
    return p;
  };
  __bf16* xbpk  = (__bf16*)alloc((size_t)Mn * Dn * sizeof(__bf16));
  __bf16* wcat  = (__bf16*)alloc((size_t)Dn * NCP * sizeof(__bf16));
  __bf16* wopk  = (__bf16*)alloc((size_t)VAL * Dn * sizeof(__bf16));
  float*  proj  = (float*) alloc((size_t)Mn * NC * sizeof(float));
  float*  qc    = (float*) alloc((size_t)Mn * KEY * sizeof(float));
  float*  kc    = (float*) alloc((size_t)Mn * KEY * sizeof(float));
  float*  vc    = (float*) alloc((size_t)Mn * VAL * sizeof(float));
  float*  beta  = (float*) alloc((size_t)Mn * Hn * sizeof(float));
  float*  gdec  = (float*) alloc((size_t)Mn * Hn * sizeof(float));
  float*  obuf  = (float*) alloc((size_t)Mn * VAL * sizeof(float));
  __bf16* obpk  = (__bf16*)alloc((size_t)Mn * VAL * sizeof(__bf16));
  (void)ws_size; (void)in_sizes; (void)n_in; (void)out_size;

  const int TB = 256;
  auto blocks = [](size_t n, int tb) { return (unsigned)((n + tb - 1) / tb); };

  // 1) pack operands into WMMA fragment layouts (bf16)
  k_pack_a<<<blocks((size_t)Mn * Dn, TB), TB, 0, stream>>>(x, xbpk, Mn, Dn);
  k_pack_wcat<<<blocks((size_t)Dn * NCP, TB), TB, 0, stream>>>(Wq, Wk, Wv, Wg, Wb, Wa, wcat);
  k_pack_wo<<<blocks((size_t)VAL * Dn, TB), TB, 0, stream>>>(Wo, wopk);

  // 2) fused projection GEMM: proj[4096][9248] = x @ [Wq|Wk|Wv|Wg|Wb|Wa]
  {
    dim3 grid(Mn / 64, NCP / 256);
    k_wmma_gemm_pk<<<grid, 256, 0, stream>>>(xbpk, wcat, proj, NC, NC, Dn, NCP / 16);
  }

  // 3) causal depthwise conv + SiLU for q/k/v
  k_conv_silu<<<blocks((size_t)Mn * (2 * KEY + VAL), TB), TB, 0, stream>>>(
      proj, cwq, cwk, cwv, qc, kc, vc);

  // 4) beta / decay gate
  k_beta_gate<<<blocks((size_t)Mn * Hn, TB), TB, 0, stream>>>(proj, Alog, dtb, beta, gdec);

  // 5) q/k L2 norm per head
  k_l2norm_qk<<<blocks((size_t)Mn * Hn * 2, TB), TB, 0, stream>>>(qc, kc);

  // 6) gated delta-rule scan (state in VGPRs)
  k_delta_scan<<<Bn * Hn, 192, 0, stream>>>(qc, kc, vc, beta, gdec, obuf);

  // 7) gated RMSNorm -> packed bf16 A-operand
  k_rms_gate<<<blocks((size_t)Mn * Hn, TB), TB, 0, stream>>>(obuf, proj, wnrm, obpk);

  // 8) output GEMM: out[4096][2048] = o_norm @ Wo
  {
    dim3 grid(Mn / 64, Dn / 256);
    k_wmma_gemm_pk<<<grid, 256, 0, stream>>>(obpk, wopk, out, Dn, Dn, VAL, Dn / 16);
  }
}
